// Adapter_MoElayer_78812649881948
// MI455X (gfx1250) — matrix operand order repaired and checked by reading, not compiled, not run on hardware
//
#include <hip/hip_runtime.h>
#include <hip/hip_bf16.h>
#include <math.h>

// ---------------- problem constants (match reference) ----------------
#define E_EXP 5
#define DIM   768
#define HID   3072
#define TOK   16384
#define EPSV  2.2204460492503131e-16f

// ---------------- types ----------------
typedef __attribute__((ext_vector_type(16))) __bf16       v16bf;
typedef __attribute__((ext_vector_type(8)))  float        v8f;
typedef __attribute__((ext_vector_type(4)))  unsigned int u32x4;
typedef __attribute__((ext_vector_type(4)))  unsigned int v4u;
typedef __attribute__((ext_vector_type(8)))  int          i32x8;
typedef __attribute__((ext_vector_type(4)))  int          i32x4;

union FragBF {
    v16bf bf;
    unsigned int u[8];
};
union FragU {
    v16bf bf;
    v4u   h[2];
};

// Tensor Data Mover available? (this toolchain lane: 6-arg clang-23 builtin)
#if defined(__gfx1250__) && __has_builtin(__builtin_amdgcn_tensor_load_to_lds) && \
    __has_builtin(__builtin_amdgcn_s_wait_tensorcnt)
#define USE_TDM 1
#else
#define USE_TDM 0
#endif

// ---------------- tiny elementwise kernels ----------------
__global__ void zero_f32_kernel(float* p, long n) {
    long i = (long)blockIdx.x * blockDim.x + threadIdx.x;
    long stride = (long)gridDim.x * blockDim.x;
    for (; i < n; i += stride) p[i] = 0.0f;
}

__global__ void f32_to_bf16_kernel(const float* __restrict__ in, __bf16* __restrict__ out, long n) {
    long i = (long)blockIdx.x * blockDim.x + threadIdx.x;
    long stride = (long)gridDim.x * blockDim.x;
    for (; i < n; i += stride) out[i] = (__bf16)in[i];
}

__global__ void final_log_kernel(float* p, long n) {
    long i = (long)blockIdx.x * blockDim.x + threadIdx.x;
    long stride = (long)gridDim.x * blockDim.x;
    for (; i < n; i += stride) {
        float v = p[i];
        p[i] = logf(v == 0.0f ? EPSV : v);
    }
}

// ---------------- LDS-tiled transpose + bf16 convert ----------------
// in: [E][R][C] fp32, out: [E][C][R] bf16   (so W^T is row-major [N][K])
__global__ __launch_bounds__(256)
void transpose_to_bf16_kernel(const float* __restrict__ in, __bf16* __restrict__ out,
                              int R, int C) {
    __shared__ float tile[32][33];
    const int e = blockIdx.z;
    const float* ine = in + (size_t)e * R * C;
    __bf16* oute = out + (size_t)e * R * C;
    const int r0 = blockIdx.y * 32;
    const int c0 = blockIdx.x * 32;
    const int tx = threadIdx.x & 31;
    const int ty = threadIdx.x >> 5;          // 0..7
#pragma unroll
    for (int i = ty; i < 32; i += 8)
        tile[i][tx] = ine[(size_t)(r0 + i) * C + (c0 + tx)];
    __syncthreads();
#pragma unroll
    for (int i = ty; i < 32; i += 8)
        oute[(size_t)(c0 + i) * R + (r0 + tx)] = (__bf16)tile[tx][i];
}

// ---------------- gating: logits = x @ w_gate ; top-2 softmax ----------------
__global__ void gating_kernel(const float* __restrict__ x,
                              const float* __restrict__ wg,   // [DIM, E]
                              float* __restrict__ gates) {    // [TOK, E]
    int lane = threadIdx.x & 31;
    int t = (int)((blockIdx.x * blockDim.x + threadIdx.x) >> 5);
    if (t >= TOK) return;

    const float* xr = x + (size_t)t * DIM;
    float a0 = 0.f, a1 = 0.f, a2 = 0.f, a3 = 0.f, a4 = 0.f;
    for (int d = lane; d < DIM; d += 32) {
        float xv = xr[d];
        const float* w = wg + (size_t)d * E_EXP;
        a0 += xv * w[0];
        a1 += xv * w[1];
        a2 += xv * w[2];
        a3 += xv * w[3];
        a4 += xv * w[4];
    }
    for (int off = 16; off > 0; off >>= 1) {
        a0 += __shfl_xor(a0, off, 32);
        a1 += __shfl_xor(a1, off, 32);
        a2 += __shfl_xor(a2, off, 32);
        a3 += __shfl_xor(a3, off, 32);
        a4 += __shfl_xor(a4, off, 32);
    }
    if (lane == 0) {
        float v[E_EXP] = {a0, a1, a2, a3, a4};
        int i0 = 0;
        for (int e = 1; e < E_EXP; ++e) if (v[e] > v[i0]) i0 = e;
        int i1 = -1;
        for (int e = 0; e < E_EXP; ++e) {
            if (e == i0) continue;
            if (i1 < 0 || v[e] > v[i1]) i1 = e;
        }
        float e1 = expf(v[i1] - v[i0]);
        float s  = 1.0f + e1;
        float g[E_EXP] = {0.f, 0.f, 0.f, 0.f, 0.f};
        g[i0] = 1.0f / s;
        g[i1] = e1 / s;
        float* gr = gates + (size_t)t * E_EXP;
        for (int e = 0; e < E_EXP; ++e) gr[e] = g[e];
    }
}

// ---------------- tiled bf16 WMMA GEMM ----------------
// C[M,N] = A[M,K] * B[K,N] (+ bias[N]); A is [M,K] bf16 row-major,
// BT is the transposed B, i.e. [N,K] bf16 row-major.
// mode 0: Hout = (bf16) gelu_erf(y);  mode 1: Cout += gate * exp(y)
#define BM 128
#define BN 128
#define BK 32

#if USE_TDM
// Issue one 2-D TDM tile load: 128 rows x 32 bf16 elements -> contiguous LDS.
__device__ __forceinline__ void tdm_load_tile_2d(unsigned lds_byte_off,
                                                 const __bf16* tile_ptr,
                                                 unsigned tensor_d0,   // row length (elems)
                                                 unsigned tensor_d1,   // #rows in tensor
                                                 unsigned stride0) {   // elems between rows
    unsigned long long ga = (unsigned long long)tile_ptr;
    u32x4 g0;
    g0[0] = 1u;                                            // count=1, user descriptor
    g0[1] = lds_byte_off;                                  // lds_addr
    g0[2] = (unsigned)(ga & 0xFFFFFFFFull);                // global_addr[31:0]
    g0[3] = (unsigned)((ga >> 32) & 0x1FFFFFFull)          // global_addr[56:32]
            | (2u << 30);                                  // type = 2 ("image")
    i32x8 g1;
    g1[0] = (int)(1u << 16);                               // data_size = 1 -> 2 bytes
    g1[1] = (int)((tensor_d0 & 0xFFFFu) << 16);            // tensor_dim0[15:0]
    g1[2] = (int)((tensor_d0 >> 16) | ((tensor_d1 & 0xFFFFu) << 16));
    g1[3] = (int)((tensor_d1 >> 16) | ((unsigned)BK << 16));  // tile_dim0 = 32
    g1[4] = (int)BM;                                       // tile_dim1 = 128, tile_dim2 = 0
    g1[5] = (int)stride0;                                  // tensor_dim0_stride[31:0]
    g1[6] = 0;
    g1[7] = 0;
    i32x4 z4 = {0, 0, 0, 0};
    i32x8 z8 = {0, 0, 0, 0, 0, 0, 0, 0};
    __builtin_amdgcn_tensor_load_to_lds(g0, g1, z4, z4, z8, 0);
}

// Fragment loads via opaque inline asm (compiler cannot see the TDM writes to
// LDS, so C-level loads get folded to undef; asm ds_loads + explicit dscnt
// wait keep the data path real).  Address math (contiguous k = 16h + s, same in A and B):
//   A (16-bit, 16x32): lane<16 -> row bytes {0..31}; lane>=16 -> +32.
//   B (stored N-major): lane<16 -> bytes {0..31}; lane>=16 -> +32.
__device__ __forceinline__ void mma_step_tdm(unsigned aBase, unsigned bBase,
                                             int lane, int wm, int wn, v8f c[2][4]) {
    const int l15 = lane & 15;
    const unsigned asel = (lane & 16) ? 32u : 0u;
    const unsigned bsel = (lane & 16) ? 32u : 0u;
    unsigned a0 = aBase + (unsigned)((wm * 32 + l15) * 64) + asel;
    unsigned a1 = a0 + 1024u;                 // +16 rows
    unsigned b0 = bBase + (unsigned)((wn * 64 + l15) * 64) + bsel;
    unsigned b1 = b0 + 1024u;                 // +16 cols
    unsigned b2 = b0 + 2048u;
    unsigned b3 = b0 + 3072u;

    FragU A0, A1, B0, B1, B2, B3;
    asm volatile(
        "ds_load_b128 %0, %12\n\t"
        "ds_load_b128 %1, %12 offset:16\n\t"
        "ds_load_b128 %2, %13\n\t"
        "ds_load_b128 %3, %13 offset:16\n\t"
        "ds_load_b128 %4, %14\n\t"
        "ds_load_b128 %5, %14 offset:16\n\t"
        "ds_load_b128 %6, %15\n\t"
        "ds_load_b128 %7, %15 offset:16\n\t"
        "ds_load_b128 %8, %16\n\t"
        "ds_load_b128 %9, %16 offset:16\n\t"
        "ds_load_b128 %10, %17\n\t"
        "ds_load_b128 %11, %17 offset:16\n\t"
        "s_wait_dscnt 0x0"
        : "=&v"(A0.h[0]), "=&v"(A0.h[1]), "=&v"(A1.h[0]), "=&v"(A1.h[1]),
          "=&v"(B0.h[0]), "=&v"(B0.h[1]), "=&v"(B1.h[0]), "=&v"(B1.h[1]),
          "=&v"(B2.h[0]), "=&v"(B2.h[1]), "=&v"(B3.h[0]), "=&v"(B3.h[1])
        : "v"(a0), "v"(a1), "v"(b0), "v"(b1), "v"(b2), "v"(b3));

    v16bf av[2] = {A0.bf, A1.bf};
    v16bf bv[4] = {B0.bf, B1.bf, B2.bf, B3.bf};
#pragma unroll
    for (int mi = 0; mi < 2; ++mi)
#pragma unroll
        for (int ni = 0; ni < 4; ++ni)
            c[mi][ni] = __builtin_amdgcn_wmma_f32_16x16x32_bf16(
                false, av[mi], false, bv[ni], (short)0, c[mi][ni], false, false);
}
#endif  // USE_TDM

// C-level fragment path (used by the non-TDM fallback, where the staging
// stores are visible to the compiler).
__device__ __forceinline__ void mma_step(const __bf16 (*Asl)[BK], const __bf16 (*Bsl)[BK],
                                         int lane, int wm, int wn, v8f c[2][4]) {
    FragBF af[2];
    const int asel = (lane < 16) ? 0 : 8;
#pragma unroll
    for (int mi = 0; mi < 2; ++mi) {
        int arow = wm * 32 + mi * 16 + (lane & 15);
        const unsigned int* ap = (const unsigned int*)&Asl[arow][0];
#pragma unroll
        for (int v = 0; v < 8; ++v)
            af[mi].u[v] = ap[v + asel];
    }
    FragBF bfr[4];
    const int bsel = (lane < 16) ? 0 : 8;
#pragma unroll
    for (int ni = 0; ni < 4; ++ni) {
        int bcol = wn * 64 + ni * 16 + (lane & 15);
        const unsigned int* bp = (const unsigned int*)&Bsl[bcol][0];
#pragma unroll
        for (int v = 0; v < 8; ++v)
            bfr[ni].u[v] = bp[v + bsel];
    }
#pragma unroll
    for (int mi = 0; mi < 2; ++mi)
#pragma unroll
        for (int ni = 0; ni < 4; ++ni)
            c[mi][ni] = __builtin_amdgcn_wmma_f32_16x16x32_bf16(
                false, af[mi].bf, false, bfr[ni].bf,
                (short)0, c[mi][ni], false, false);
}

__global__ __launch_bounds__(256)
void moe_gemm_wmma_kernel(const __bf16* __restrict__ A,
                          const __bf16* __restrict__ BT,    // [N,K] row-major
                          const float*  __restrict__ bias,
                          __bf16*       __restrict__ Hout,
                          float*        __restrict__ Cout,
                          const float*  __restrict__ gates,
                          int M, int N, int Kdim, int expert, int mode) {
    // Single LDS object -> guaranteed offset 0.
    // Layout: A0 @ 0, A1 @ 8192, B0 @ 16384, B1 @ 24576 (bytes), 8KB tiles.
    __shared__ __bf16 smem[4 * BM * BK];

    const int tid  = threadIdx.x;
    const int lane = tid & 31;
    const int wave = tid >> 5;
    const int wm   = wave & 3;        // M offset wm*32
    const int wn   = wave >> 2;       // N offset wn*64

    const int n0 = blockIdx.x * BN;
    const int m0 = blockIdx.y * BM;
    const int nk = Kdim / BK;

    v8f c[2][4];
    const v8f z8 = {0.f, 0.f, 0.f, 0.f, 0.f, 0.f, 0.f, 0.f};
#pragma unroll
    for (int mi = 0; mi < 2; ++mi)
#pragma unroll
        for (int ni = 0; ni < 4; ++ni) c[mi][ni] = z8;

#if USE_TDM
    // Anchor the LDS allocation (never executes; Kdim is a runtime arg).
    if (Kdim == 0) ((volatile __bf16*)smem)[tid] = (__bf16)0.f;

    const unsigned offA0 = 0u;
    const unsigned offA1 = (unsigned)(BM * BK * 2);
    const unsigned offB0 = (unsigned)(2 * BM * BK * 2);
    const unsigned offB1 = (unsigned)(3 * BM * BK * 2);

    if (wave == 0) {  // one issuing wave; TDM in-order per wave
        tdm_load_tile_2d(offA0, A  + (size_t)m0 * Kdim, (unsigned)Kdim, (unsigned)M, (unsigned)Kdim);
        tdm_load_tile_2d(offB0, BT + (size_t)n0 * Kdim, (unsigned)Kdim, (unsigned)N, (unsigned)Kdim);
    }
    for (int kk = 0; kk < nk; ++kk) {
        const int cur = kk & 1;
        if (wave == 0) {
            if (kk + 1 < nk) {
                const int k1 = (kk + 1) * BK;
                // prefetch next tiles into the other buffer, then wait for current pair
                tdm_load_tile_2d(cur ? offA0 : offA1, A  + (size_t)m0 * Kdim + k1,
                                 (unsigned)Kdim, (unsigned)M, (unsigned)Kdim);
                tdm_load_tile_2d(cur ? offB0 : offB1, BT + (size_t)n0 * Kdim + k1,
                                 (unsigned)Kdim, (unsigned)N, (unsigned)Kdim);
                __builtin_amdgcn_s_wait_tensorcnt(2);
            } else {
                __builtin_amdgcn_s_wait_tensorcnt(0);
            }
        }
        __syncthreads();                    // current tiles visible to all waves
        mma_step_tdm(cur ? offA1 : offA0, cur ? offB1 : offB0, lane, wm, wn, c);
        __syncthreads();                    // all waves done before buffer reuse
    }
#else
    // fallback: synchronous staging (both tiles are [128 rows x 32 bf16])
    __bf16 (*As0)[BK] = (__bf16 (*)[BK])(smem);
    __bf16 (*Bs0)[BK] = (__bf16 (*)[BK])(smem + 2 * BM * BK);
    for (int kk = 0; kk < nk; ++kk) {
        const int k0 = kk * BK;
#pragma unroll
        for (int i = 0; i < 2; ++i) {
            int idx = tid + i * 256;      // 0..511 uint4
            int row = idx >> 2;
            int c4  = idx & 3;
            const uint4* srcA = (const uint4*)(A + (size_t)(m0 + row) * Kdim + k0) + c4;
            *((uint4*)&As0[row][c4 * 8]) = *srcA;
            const uint4* srcB = (const uint4*)(BT + (size_t)(n0 + row) * Kdim + k0) + c4;
            *((uint4*)&Bs0[row][c4 * 8]) = *srcB;
        }
        if (k0 + BK < Kdim) {
            __builtin_prefetch(A  + (size_t)(m0 + (tid & 127)) * Kdim + (k0 + BK), 0, 0);
            __builtin_prefetch(BT + (size_t)(n0 + (tid & 127)) * Kdim + (k0 + BK), 0, 0);
        }
        __syncthreads();
        mma_step(As0, Bs0, lane, wm, wn, c);
        __syncthreads();
    }
#endif

    // ---- epilogue ----
    const int rbase = ((lane >= 16) ? 8 : 0);
    const int ncol_lane = lane & 15;
#pragma unroll
    for (int mi = 0; mi < 2; ++mi) {
#pragma unroll
        for (int ni = 0; ni < 4; ++ni) {
            int coln = n0 + wn * 64 + ni * 16 + ncol_lane;
            float bv = bias[coln];
#pragma unroll
            for (int r = 0; r < 8; ++r) {
                int row = m0 + wm * 32 + mi * 16 + r + rbase;
                float y = c[mi][ni][r] + bv;
                if (mode == 0) {
                    float g = 0.5f * y * (1.0f + erff(y * 0.70710678118654752f));
                    Hout[(size_t)row * N + coln] = (__bf16)g;
                } else {
                    float gate = gates[(size_t)row * E_EXP + expert];
                    float* dst = Cout + (size_t)row * N + coln;
                    *dst = *dst + gate * expf(y);
                }
            }
        }
    }
}

// ---------------- host launch ----------------
extern "C" void kernel_launch(void* const* d_in, const int* in_sizes, int n_in,
                              void* d_out, int out_size, void* d_ws, size_t ws_size,
                              hipStream_t stream) {
    (void)in_sizes; (void)n_in; (void)out_size; (void)ws_size;

    const float* x      = (const float*)d_in[0];   // [TOK, DIM]
    const float* w_gate = (const float*)d_in[1];   // [DIM, E]
    const float* W1     = (const float*)d_in[2];   // [E, DIM, HID]
    const float* b1     = (const float*)d_in[3];   // [E, HID]
    const float* W2     = (const float*)d_in[4];   // [E, HID, DIM]
    const float* b2     = (const float*)d_in[5];   // [E, DIM]
    float* out = (float*)d_out;                    // [TOK, DIM]

    // workspace layout (all multiples of 256 bytes)
    char* p = (char*)d_ws;
    __bf16* xb   = (__bf16*)p; p += (size_t)TOK * DIM * sizeof(__bf16);
    __bf16* w1t  = (__bf16*)p; p += (size_t)E_EXP * DIM * HID * sizeof(__bf16);   // [E][HID][DIM]
    __bf16* w2t  = (__bf16*)p; p += (size_t)E_EXP * HID * DIM * sizeof(__bf16);   // [E][DIM][HID]
    __bf16* hb   = (__bf16*)p; p += (size_t)TOK * HID * sizeof(__bf16);           // [TOK][HID]
    float*  gts  = (float*)p;  p += (size_t)TOK * E_EXP * sizeof(float);

    const long nOut = (long)TOK * DIM;

    // 1) zero the accumulator (d_out doubles as acc)
    zero_f32_kernel<<<2048, 256, 0, stream>>>(out, nOut);

    // 2) bf16 conversions; weights also transposed so GEMM B-tiles are TDM-friendly
    f32_to_bf16_kernel<<<2048, 256, 0, stream>>>(x, xb, (long)TOK * DIM);
    {
        dim3 g1(HID / 32, DIM / 32, E_EXP);   // W1 [DIM][HID] -> [HID][DIM]
        transpose_to_bf16_kernel<<<g1, 256, 0, stream>>>(W1, w1t, DIM, HID);
        dim3 g2(DIM / 32, HID / 32, E_EXP);   // W2 [HID][DIM] -> [DIM][HID]
        transpose_to_bf16_kernel<<<g2, 256, 0, stream>>>(W2, w2t, HID, DIM);
    }

    // 3) gating: one wave per token
    gating_kernel<<<TOK / 8, 256, 0, stream>>>(x, w_gate, gts);

    // 4) per-expert GEMM pipeline (sequential launches -> race-free RMW on out)
    dim3 blk(256);
    dim3 grid1(HID / BN, TOK / BM);   // 24 x 128
    dim3 grid2(DIM / BN, TOK / BM);   // 6  x 128
    for (int e = 0; e < E_EXP; ++e) {
        moe_gemm_wmma_kernel<<<grid1, blk, 0, stream>>>(
            xb, w1t + (size_t)e * DIM * HID, b1 + (size_t)e * HID,
            hb, nullptr, nullptr,
            TOK, HID, DIM, e, /*mode=*/0);
        moe_gemm_wmma_kernel<<<grid2, blk, 0, stream>>>(
            hb, w2t + (size_t)e * HID * DIM, b2 + (size_t)e * DIM,
            nullptr, out, gts,
            TOK, DIM, HID, e, /*mode=*/1);
    }

    // 5) out = log(out == 0 ? eps : out)
    final_log_kernel<<<2048, 256, 0, stream>>>(out, nOut);
}
